// Encoder_3049426780874
// MI455X (gfx1250) — matrix operand order repaired and checked by reading, not compiled, not run on hardware
//
#include <hip/hip_runtime.h>
#include <hip/hip_bf16.h>

// ---------------------------------------------------------------------------
// Types
// ---------------------------------------------------------------------------
typedef __attribute__((ext_vector_type(16))) __bf16 v16bf;
typedef __attribute__((ext_vector_type(8)))  __bf16 bf8;
typedef __attribute__((ext_vector_type(4)))  __bf16 bf4;
typedef __attribute__((ext_vector_type(8)))  float  v8f;
typedef __attribute__((ext_vector_type(4)))  float  f4;

#define K_BRANCH 8
#define BATCH    4096
#define IN_DIM   1024
#define HID_DIM  2048
#define OUT_DIM  256
#define NPAIR    56

// GEMM tiling: block computes 128x128, 8 waves of 32x64, BK=32
#define BM 128
#define BN 128
#define BK 32
#define LDT 40   // padded tile row (elements); 80B rows -> 16B-aligned chunks

// Async global->LDS staging (CDNA5): use if the toolchain exposes it.
#if defined(__HIP_DEVICE_COMPILE__) && \
    __has_builtin(__builtin_amdgcn_global_load_async_to_lds_b128) && \
    __has_builtin(__builtin_amdgcn_s_wait_asynccnt)
#define USE_ASYNC_LDS 1
#else
#define USE_ASYNC_LDS 0
#endif

#if USE_ASYNC_LDS
typedef int v4i_vs __attribute__((__vector_size__(4 * sizeof(int))));
#define GPTR(p) ((__attribute__((address_space(1))) v4i_vs*)(unsigned long long)(p))
#define LPTR(p) ((__attribute__((address_space(3))) v4i_vs*)(unsigned)(unsigned long long)(p))
#endif

// ---------------------------------------------------------------------------
// fp32 -> bf16 convert (vectorized, grid-stride; n multiple of 4)
// ---------------------------------------------------------------------------
__global__ __launch_bounds__(256) void cvt_f32_bf16(const float* __restrict__ src,
                                                    __bf16* __restrict__ dst,
                                                    long long n4) {
  long long i = (long long)blockIdx.x * blockDim.x + threadIdx.x;
  long long stride = (long long)gridDim.x * blockDim.x;
  for (; i < n4; i += stride) {
    f4 v = *(const f4*)(src + 4 * i);
    bf4 o;
    o.x = (__bf16)v.x; o.y = (__bf16)v.y; o.z = (__bf16)v.z; o.w = (__bf16)v.w;
    *(bf4*)(dst + 4 * i) = o;
  }
}

// ---------------------------------------------------------------------------
// Tiled fp32->bf16 transpose: dst[C,R] = src[R,C]^T, batched over g.
// grid = (C/32, R/32, G), block = 256 (32x8)
// ---------------------------------------------------------------------------
__global__ __launch_bounds__(256)
void transpose_cvt(const float* __restrict__ src, long long sSrc,
                   __bf16* __restrict__ dst, long long sDst, int R, int C) {
  __shared__ float t[32][33];
  const int g = blockIdx.z;
  src += (long long)g * sSrc;
  dst += (long long)g * sDst;
  const int r0 = blockIdx.y * 32;
  const int c0 = blockIdx.x * 32;
  const int tx = threadIdx.x & 31;
  const int ty = threadIdx.x >> 5;  // 0..7
#pragma unroll
  for (int i = 0; i < 4; ++i) {
    int r = ty + i * 8;
    t[r][tx] = src[(long long)(r0 + r) * C + c0 + tx];
  }
  __syncthreads();
#pragma unroll
  for (int i = 0; i < 4; ++i) {
    int r = ty + i * 8;  // row in dst tile == col in src tile
    dst[(long long)(c0 + r) * R + r0 + tx] = (__bf16)t[tx][r];
  }
}

// ---------------------------------------------------------------------------
// Fragment load: 16 bf16 as two contiguous 16B runs (maps to 2x ds_load_b128)
// ---------------------------------------------------------------------------
__device__ __forceinline__ v16bf load_frag(const __bf16* p0, const __bf16* p1) {
  bf8 lo = *(const bf8*)p0;
  bf8 hi = *(const bf8*)p1;
  v16bf r;
#pragma unroll
  for (int u = 0; u < 8; ++u) { r[u] = lo[u]; r[8 + u] = hi[u]; }
  return r;
}

// ---------------------------------------------------------------------------
// Tiled bf16 WMMA GEMM with TRANSPOSED weights:
//   D[g] = act(A[g] @ Wt[g]^T + bias[g]) -> bf16
//   A : [M, lda]  bf16 row-major, group stride sA (0 => shared)
//   Wt: [N, ldw]  bf16 row-major (weight already transposed: Wt[n][k])
//   Requires M%128==0, N%128==0, Kd%32==0
//   grid = (N/128, M/128, groups), block = 256 (8 waves: 4x2 wave tiles)
//   Double-buffered LDS; async global->LDS staging when available.
// ---------------------------------------------------------------------------
__global__ __launch_bounds__(256)
void gemm_bf16_wmma(const __bf16* __restrict__ A, long long sA, int lda,
                    const __bf16* __restrict__ Wt, long long sW, int ldw,
                    const float*  __restrict__ bias, long long sBias,
                    __bf16* __restrict__ D, long long sD, int ldd,
                    int Kd, int relu) {
  __shared__ __bf16 shA[2][BM * LDT];
  __shared__ __bf16 shB[2][BN * LDT];

  const int g = blockIdx.z;
  A  += (long long)g * sA;
  Wt += (long long)g * sW;
  D  += (long long)g * sD;
  const float* bptr = bias ? (bias + (long long)g * sBias) : (const float*)nullptr;

  const int tid  = threadIdx.x;
  const int lane = tid & 31;
  const int wave = tid >> 5;
  const int wm   = wave >> 1;  // 0..3 -> 32-row strip
  const int wn   = wave & 1;   // 0..1 -> 64-col strip

  const int row0 = blockIdx.y * BM;
  const int col0 = blockIdx.x * BN;

  v8f acc[2][4];
#pragma unroll
  for (int m = 0; m < 2; ++m)
#pragma unroll
    for (int n = 0; n < 4; ++n)
#pragma unroll
      for (int e = 0; e < 8; ++e) acc[m][n][e] = 0.0f;

  // per-lane fragment geometry (wave32 ISA layouts)
  const int m_lane  = lane & 15;
  const int k_halfA = (lane >> 4) << 3;   // A: 0 or 8
  const int k_halfB = (lane >> 4) << 4;   // B: 0 or 16

  // stage one BKx(128+128) tile pair into buffer `buf` (4 x 16B per thread)
  auto stage = [&](int buf, int kk) {
#pragma unroll
    for (int c = 0; c < 2; ++c) {
      int idx = tid + c * 256;          // 0..511
      int r   = idx >> 2;               // 0..127
      int cc  = (idx & 3) * 8;          // 0,8,16,24
      const __bf16* ga = &A [(long long)(row0 + r) * lda + kk + cc];
      const __bf16* gb = &Wt[(long long)(col0 + r) * ldw + kk + cc];
      __bf16* la = &shA[buf][r * LDT + cc];
      __bf16* lb = &shB[buf][r * LDT + cc];
#if USE_ASYNC_LDS
      __builtin_amdgcn_global_load_async_to_lds_b128(GPTR(ga), LPTR(la), 0, 0);
      __builtin_amdgcn_global_load_async_to_lds_b128(GPTR(gb), LPTR(lb), 0, 0);
#else
      *(uint4*)la = *(const uint4*)ga;
      *(uint4*)lb = *(const uint4*)gb;
#endif
    }
  };

  const int nk = Kd / BK;
  stage(0, 0);

  for (int kt = 0; kt < nk; ++kt) {
    const int buf = kt & 1;
    const bool more = (kt + 1) < nk;
    if (more) stage(buf ^ 1, (kt + 1) * BK);
#if USE_ASYNC_LDS
    if (more) __builtin_amdgcn_s_wait_asynccnt(4);  // retire current buffer
    else      __builtin_amdgcn_s_wait_asynccnt(0);
#endif
    __syncthreads();  // all waves' staging of `buf` visible

    // A fragments: two contiguous 8-elem runs per lane
    v16bf af[2];
#pragma unroll
    for (int ms = 0; ms < 2; ++ms) {
      const __bf16* base = &shA[buf][(wm * 32 + ms * 16 + m_lane) * LDT];
      af[ms] = load_frag(base + k_halfA, base + 16 + k_halfA);
    }
    // B fragments: two contiguous 8-elem runs per lane, same k order as A (Wt is [N,K])
    v16bf bfr[4];
#pragma unroll
    for (int ns = 0; ns < 4; ++ns) {
      const __bf16* base = &shB[buf][(wn * 64 + ns * 16 + m_lane) * LDT + k_halfA];
      bfr[ns] = load_frag(base, base + 16);
    }
#pragma unroll
    for (int ms = 0; ms < 2; ++ms)
#pragma unroll
      for (int ns = 0; ns < 4; ++ns)
        acc[ms][ns] = __builtin_amdgcn_wmma_f32_16x16x32_bf16(
            false, af[ms], false, bfr[ns], (short)0, acc[ms][ns], false, false);

    __syncthreads();  // readers done before buffer reuse
  }

  // epilogue: bias + optional ReLU, store bf16
  const int n_lane = lane & 15;
  const int m_off  = (lane >> 4) << 3;  // 0 or 8
#pragma unroll
  for (int ms = 0; ms < 2; ++ms) {
#pragma unroll
    for (int ns = 0; ns < 4; ++ns) {
      int col = col0 + wn * 64 + ns * 16 + n_lane;
      float bv = bptr ? bptr[col] : 0.0f;
#pragma unroll
      for (int r = 0; r < 8; ++r) {
        float v = acc[ms][ns][r] + bv;
        if (relu) v = fmaxf(v, 0.0f);
        int row = row0 + wm * 32 + ms * 16 + m_off + r;
        D[(long long)row * ldd + col] = (__bf16)v;
      }
    }
  }
}

// ---------------------------------------------------------------------------
// Pairwise "on" head: out[b,p] = bon2 + sum_h relu(Aon[i,b,h]+Bon[j,b,h]+bon1[h])*Won2[h]
// one wave per (p,b) row; 8 rows per block
// ---------------------------------------------------------------------------
__global__ __launch_bounds__(256)
void reduce_on(const __bf16* __restrict__ Aon, const __bf16* __restrict__ Bon,
               const float* __restrict__ bon1, const float* __restrict__ Won2,
               const float* __restrict__ bon2, float* __restrict__ out) {
  __shared__ float s_w[HID_DIM];
  __shared__ float s_b[HID_DIM];
  for (int i = threadIdx.x; i < HID_DIM; i += 256) {
    s_w[i] = Won2[i];
    s_b[i] = bon1[i];
  }
  __syncthreads();

  const int wave = threadIdx.x >> 5;
  const int lane = threadIdx.x & 31;
  long long row = (long long)blockIdx.x * 8 + wave;  // [0, 56*4096)
  const int b = (int)(row & (BATCH - 1));
  const int p = (int)(row >> 12);
  const int i_ = p / 7;
  const int r_ = p % 7;
  const int j_ = r_ + (r_ >= i_ ? 1 : 0);

  const __bf16* pa = Aon + ((long long)i_ * BATCH + b) * HID_DIM;
  const __bf16* pb = Bon + ((long long)j_ * BATCH + b) * HID_DIM;

  float acc = 0.0f;
#pragma unroll
  for (int it = 0; it < HID_DIM / (32 * 8); ++it) {
    int h0 = (it * 32 + lane) * 8;
    bf8 va = *(const bf8*)&pa[h0];
    bf8 vb = *(const bf8*)&pb[h0];
#pragma unroll
    for (int u = 0; u < 8; ++u) {
      float x = (float)va[u] + (float)vb[u] + s_b[h0 + u];
      x = fmaxf(x, 0.0f);
      acc = fmaf(x, s_w[h0 + u], acc);
    }
  }
#pragma unroll
  for (int m = 16; m > 0; m >>= 1) acc += __shfl_xor(acc, m, 32);
  if (lane == 0) out[(long long)b * 64 + p] = acc + bon2[0];
}

// ---------------------------------------------------------------------------
// Clear heads: out[b,56+k] = bc2 + sum_h relu(Cc[k,b,h]+bc1[h])*Wc2[h]
// ---------------------------------------------------------------------------
__global__ __launch_bounds__(256)
void reduce_clear(const __bf16* __restrict__ Cc, const float* __restrict__ bc1,
                  const float* __restrict__ Wc2, const float* __restrict__ bc2,
                  float* __restrict__ out) {
  __shared__ float s_w[HID_DIM];
  __shared__ float s_b[HID_DIM];
  for (int i = threadIdx.x; i < HID_DIM; i += 256) {
    s_w[i] = Wc2[i];
    s_b[i] = bc1[i];
  }
  __syncthreads();

  const int wave = threadIdx.x >> 5;
  const int lane = threadIdx.x & 31;
  long long row = (long long)blockIdx.x * 8 + wave;  // [0, 8*4096)
  const int b = (int)(row & (BATCH - 1));
  const int k = (int)(row >> 12);

  const __bf16* pc = Cc + ((long long)k * BATCH + b) * HID_DIM;

  float acc = 0.0f;
#pragma unroll
  for (int it = 0; it < HID_DIM / (32 * 8); ++it) {
    int h0 = (it * 32 + lane) * 8;
    bf8 vc = *(const bf8*)&pc[h0];
#pragma unroll
    for (int u = 0; u < 8; ++u) {
      float x = (float)vc[u] + s_b[h0 + u];
      x = fmaxf(x, 0.0f);
      acc = fmaf(x, s_w[h0 + u], acc);
    }
  }
#pragma unroll
  for (int m = 16; m > 0; m >>= 1) acc += __shfl_xor(acc, m, 32);
  if (lane == 0) out[(long long)b * 64 + 56 + k] = acc + bc2[0];
}

// ---------------------------------------------------------------------------
// Host-side orchestration
// ---------------------------------------------------------------------------
extern "C" void kernel_launch(void* const* d_in, const int* in_sizes, int n_in,
                              void* d_out, int out_size, void* d_ws, size_t ws_size,
                              hipStream_t stream) {
  const float* x    = (const float*)d_in[0];
  const float* W1   = (const float*)d_in[1];
  const float* b1   = (const float*)d_in[2];
  const float* W2   = (const float*)d_in[3];
  const float* b2   = (const float*)d_in[4];
  const float* Won1 = (const float*)d_in[5];
  const float* bon1 = (const float*)d_in[6];
  const float* Won2 = (const float*)d_in[7];
  const float* bon2 = (const float*)d_in[8];
  const float* Wc1  = (const float*)d_in[9];
  const float* bc1  = (const float*)d_in[10];
  const float* Wc2  = (const float*)d_in[11];
  const float* bc2  = (const float*)d_in[12];
  float* out = (float*)d_out;

  // workspace layout
  char* ws = (char*)d_ws;
  size_t off = 0;
  auto take = [&](size_t bytes) { char* p = ws + off; off += (bytes + 255) & ~(size_t)255; return p; };

  __bf16* Xbf    = (__bf16*)take((size_t)BATCH * IN_DIM * 2);
  __bf16* W1t    = (__bf16*)take((size_t)K_BRANCH * IN_DIM * HID_DIM * 2);   // [8, 2048, 1024]
  __bf16* W2t    = (__bf16*)take((size_t)K_BRANCH * HID_DIM * OUT_DIM * 2);  // [8, 256, 2048]
  __bf16* Won1t  = (__bf16*)take((size_t)HID_DIM * 2 * OUT_DIM * 2);         // [2048, 512]
  __bf16* Wc1t   = (__bf16*)take((size_t)HID_DIM * OUT_DIM * 2);             // [2048, 256]
  __bf16* Hbf    = (__bf16*)take((size_t)K_BRANCH * BATCH * HID_DIM * 2);    // reused for Cc
  __bf16* Ebf    = (__bf16*)take((size_t)K_BRANCH * BATCH * OUT_DIM * 2);
  __bf16* Aon    = (__bf16*)take((size_t)K_BRANCH * BATCH * HID_DIM * 2);
  __bf16* Bon    = (__bf16*)take((size_t)K_BRANCH * BATCH * HID_DIM * 2);
  (void)ws_size; (void)n_in; (void)in_sizes; (void)out_size;

  // convert X; transpose+convert all weight matrices (B operands want [N,K])
  cvt_f32_bf16<<<1024, 256, 0, stream>>>(x, Xbf, (long long)BATCH * IN_DIM / 4);
  transpose_cvt<<<dim3(HID_DIM / 32, IN_DIM / 32, K_BRANCH), 256, 0, stream>>>(
      W1, (long long)IN_DIM * HID_DIM, W1t, (long long)IN_DIM * HID_DIM, IN_DIM, HID_DIM);
  transpose_cvt<<<dim3(OUT_DIM / 32, HID_DIM / 32, K_BRANCH), 256, 0, stream>>>(
      W2, (long long)HID_DIM * OUT_DIM, W2t, (long long)HID_DIM * OUT_DIM, HID_DIM, OUT_DIM);
  transpose_cvt<<<dim3(HID_DIM / 32, (2 * OUT_DIM) / 32, 1), 256, 0, stream>>>(
      Won1, 0LL, Won1t, 0LL, 2 * OUT_DIM, HID_DIM);
  transpose_cvt<<<dim3(HID_DIM / 32, OUT_DIM / 32, 1), 256, 0, stream>>>(
      Wc1, 0LL, Wc1t, 0LL, OUT_DIM, HID_DIM);

  // 1) H = relu(X @ W1_k + b1_k)   [8, 4096, 2048]
  gemm_bf16_wmma<<<dim3(HID_DIM / BN, BATCH / BM, K_BRANCH), 256, 0, stream>>>(
      Xbf, 0LL, IN_DIM,
      W1t, (long long)IN_DIM * HID_DIM, IN_DIM,
      b1, (long long)HID_DIM,
      Hbf, (long long)BATCH * HID_DIM, HID_DIM,
      IN_DIM, 1);

  // 2) emb = H_k @ W2_k + b2_k     [8, 4096, 256]
  gemm_bf16_wmma<<<dim3(OUT_DIM / BN, BATCH / BM, K_BRANCH), 256, 0, stream>>>(
      Hbf, (long long)BATCH * HID_DIM, HID_DIM,
      W2t, (long long)HID_DIM * OUT_DIM, HID_DIM,
      b2, (long long)OUT_DIM,
      Ebf, (long long)BATCH * OUT_DIM, OUT_DIM,
      HID_DIM, 0);

  // 3) Aon_k = emb_k @ Won1[:256]  (no bias/act)
  gemm_bf16_wmma<<<dim3(HID_DIM / BN, BATCH / BM, K_BRANCH), 256, 0, stream>>>(
      Ebf, (long long)BATCH * OUT_DIM, OUT_DIM,
      Won1t, 0LL, 2 * OUT_DIM,
      (const float*)nullptr, 0LL,
      Aon, (long long)BATCH * HID_DIM, HID_DIM,
      OUT_DIM, 0);

  // 4) Bon_k = emb_k @ Won1[256:]
  gemm_bf16_wmma<<<dim3(HID_DIM / BN, BATCH / BM, K_BRANCH), 256, 0, stream>>>(
      Ebf, (long long)BATCH * OUT_DIM, OUT_DIM,
      Won1t + OUT_DIM, 0LL, 2 * OUT_DIM,
      (const float*)nullptr, 0LL,
      Bon, (long long)BATCH * HID_DIM, HID_DIM,
      OUT_DIM, 0);

  // 5) Cc_k = emb_k @ Wc1  (reuse H buffer)
  gemm_bf16_wmma<<<dim3(HID_DIM / BN, BATCH / BM, K_BRANCH), 256, 0, stream>>>(
      Ebf, (long long)BATCH * OUT_DIM, OUT_DIM,
      Wc1t, 0LL, OUT_DIM,
      (const float*)nullptr, 0LL,
      Hbf, (long long)BATCH * HID_DIM, HID_DIM,
      OUT_DIM, 0);

  // 6) pairwise on-head -> out[:, 0:56]
  reduce_on<<<(NPAIR * BATCH) / 8, 256, 0, stream>>>(Aon, Bon, bon1, Won2, bon2, out);

  // 7) clear heads -> out[:, 56:64]
  reduce_clear<<<(K_BRANCH * BATCH) / 8, 256, 0, stream>>>(Hbf, bc1, Wc2, bc2, out);
}
